// Mamba3Block_71768903516850
// MI455X (gfx1250) — matrix operand order repaired and checked by reading, not compiled, not run on hardware
//
#include <hip/hip_runtime.h>
#include <hip/hip_bf16.h>
#include <math.h>

// ---------------- model constants ----------------
#define LSEQ 1024
#define DMODEL 512
#define NSTATE 64
#define PHEAD 64
#define RRANK 2
#define CHUNK 128
#define NCHUNK 8
#define HHEADS 16
#define DPROJ 2306          // 1024 z + 1024 x + 128 B + 128 C + 1 dt + 1 lam
#define OFF_Z 0
#define OFF_X 1024
#define OFF_B 2048
#define OFF_C 2176
#define OFF_DT 2304
#define OFF_LAM 2305

typedef __attribute__((ext_vector_type(16))) __bf16 v16bf;
typedef __attribute__((ext_vector_type(8)))  float  v8f;
typedef __attribute__((ext_vector_type(4)))  unsigned u32x4;
typedef __attribute__((ext_vector_type(8)))  unsigned u32x8;

__device__ __forceinline__ __bf16 f2bf(float f) {
  union { float f; unsigned u; } v; v.f = f;
  unsigned r = (v.u + 0x7FFFu + ((v.u >> 16) & 1u)) >> 16;   // RNE
  union { unsigned short s; __bf16 b; } o; o.s = (unsigned short)r;
  return o.b;
}

// ---------------------------------------------------------------------
// Tensor Data Mover: 2-D f32 tile  global -> LDS  (D# groups 0+1 only,
// per CDNA5 ISA; VADDR2/3 NULL => up-to-2D form).
// Issued per-wave; completion via TENSORcnt.
// ---------------------------------------------------------------------
__device__ __forceinline__ void tdm_load_2d_f32(unsigned lds_byte_off,
                                                const void* gsrc,
                                                unsigned tile_d0, unsigned tile_d1,
                                                unsigned long long dim0_stride_elems,
                                                unsigned tensor_d0, unsigned tensor_d1) {
  unsigned long long ga = (unsigned long long)gsrc;
  u32x4 g0;
  g0[0] = 1u;                                                  // count=1, user D#
  g0[1] = lds_byte_off;                                        // lds_addr
  g0[2] = (unsigned)ga;                                        // global_addr[31:0]
  g0[3] = (unsigned)((ga >> 32) & 0x01FFFFFFull) | (2u << 30); // addr[56:32] | type=2
  u32x8 g1;
  g1[0] = 2u << 16;                                            // wg_mask=0, data_size=4B
  g1[1] = (tensor_d0 & 0xFFFFu) << 16;                         // abar=0 | tensor_dim0 lo16
  g1[2] = ((tensor_d0 >> 16) & 0xFFFFu) | ((tensor_d1 & 0xFFFFu) << 16);
  g1[3] = ((tensor_d1 >> 16) & 0xFFFFu) | ((tile_d0 & 0xFFFFu) << 16);
  g1[4] = (tile_d1 & 0xFFFFu);                                 // tile_dim1 | tile_dim2=0
  g1[5] = (unsigned)dim0_stride_elems;                         // tensor_dim0_stride lo32
  g1[6] = (unsigned)((dim0_stride_elems >> 32) & 0xFFFFull);   // stride hi | dim1_stride lo=0
  g1[7] = 0u;                                                  // dim1_stride hi
  asm volatile("tensor_load_to_lds %0, %1" :: "s"(g0), "s"(g1) : "memory");
}

// ---------------- workspace layout (bytes) ----------------
constexpr size_t wal(size_t x) { return (x + 255) & ~size_t(255); }
constexpr size_t WS_PROJ  = 0;                                        // L*DPROJ f32
constexpr size_t WS_CSDT  = wal(WS_PROJ  + (size_t)LSEQ*DPROJ*4);     // L f32  (global cumsum of dt)
constexpr size_t WS_LACS  = wal(WS_CSDT  + LSEQ*4);                   // L f32  (per-chunk cumsum of log_alpha)
constexpr size_t WS_USC   = wal(WS_LACS  + LSEQ*4);                   // L f32  (dt*lam)
constexpr size_t WS_THETA = wal(WS_USC   + LSEQ*4);                   // 32 f32
constexpr size_t WS_DECC  = wal(WS_THETA + 32*4);                     // 8 f32
constexpr size_t WS_BROT  = wal(WS_DECC  + 8*4);                      // L*128 f32
constexpr size_t WS_CROT  = wal(WS_BROT  + (size_t)LSEQ*128*4);       // L*128 f32
constexpr size_t WS_XW    = wal(WS_CROT  + (size_t)LSEQ*128*4);       // L*H*128 f32
constexpr size_t WS_YDIAG = wal(WS_XW    + (size_t)LSEQ*HHEADS*128*4);// L*H*128 f32
constexpr size_t WS_HLAST = wal(WS_YDIAG + (size_t)LSEQ*HHEADS*128*4);// 8*16*4096 f32
constexpr size_t WS_HPREV = wal(WS_HLAST + (size_t)NCHUNK*HHEADS*4096*4);
constexpr size_t WS_YCORE = wal(WS_HPREV + (size_t)NCHUNK*HHEADS*4096*4); // L*1024 f32

// =====================================================================
// Generic bf16-WMMA GEMM:  C[M,N] = A[M,K] * Bn[N,K]^T (+ bias[N])
// 256 threads, 128x128 output tile, K stepped by 32.
// =====================================================================
__global__ void gemm_bf16_wmma(const float* __restrict__ A,
                               const float* __restrict__ Bn,
                               const float* __restrict__ bias,
                               float* __restrict__ C,
                               int M, int N, int K) {
  __shared__ __bf16 As[128][40];   // [m][k], padded
  __shared__ __bf16 Bs[128][40];   // [n][k], padded

  const int tid  = threadIdx.x;
  const int lane = tid & 31;
  const int wv   = tid >> 5;           // 8 waves: one 16-row M strip each
  const int mrow = lane & 15;
  const int kh   = lane >> 4;

  const int n0 = blockIdx.x * 128;
  const int m0 = blockIdx.y * 128;

  v8f acc[8] = {};

  for (int k0 = 0; k0 < K; k0 += 32) {
    for (int idx = tid; idx < 128 * 32; idx += 256) {
      int r = idx >> 5, kk = idx & 31;
      int m = m0 + r;
      int n = n0 + r;
      float av = (m < M && (k0 + kk) < K) ? A[(size_t)m * K + k0 + kk] : 0.f;
      float bv = (n < N && (k0 + kk) < K) ? Bn[(size_t)n * K + k0 + kk] : 0.f;
      As[r][kk] = f2bf(av);
      Bs[r][kk] = f2bf(bv);
    }
    __syncthreads();

    v16bf a;
#pragma unroll
    for (int s = 0; s < 16; ++s) {
      int kk = (s < 8) ? (kh * 8 + s) : (16 + kh * 8 + (s - 8));
      a[s] = As[wv * 16 + mrow][kk];
    }
#pragma unroll
    for (int t = 0; t < 8; ++t) {
      v16bf b;
#pragma unroll
      for (int s = 0; s < 16; ++s)
        b[s] = Bs[t * 16 + (lane & 15)][(s < 8) ? (kh * 8 + s) : (16 + kh * 8 + (s - 8))];
      acc[t] = __builtin_amdgcn_wmma_f32_16x16x32_bf16(
          false, a, false, b, (short)0, acc[t], false, false);
    }
    __syncthreads();
  }

#pragma unroll
  for (int t = 0; t < 8; ++t)
#pragma unroll
    for (int v = 0; v < 8; ++v) {
      int m = m0 + wv * 16 + v + 8 * kh;
      int n = n0 + t * 16 + (lane & 15);
      if (m < M && n < N)
        C[(size_t)m * N + n] = acc[t][v] + (bias ? bias[n] : 0.f);
    }
}

// =====================================================================
// Serial scalar prep: softplus(dt), sigmoid(lam), cumsums, chunk decays
// =====================================================================
__global__ void prep_scalars(const float* __restrict__ proj,
                             const float* __restrict__ A_log,
                             const float* __restrict__ theta_log,
                             float* __restrict__ csdt, float* __restrict__ lacs,
                             float* __restrict__ uscale, float* __restrict__ theta,
                             float* __restrict__ decayc) {
  int t = threadIdx.x;
  if (t < 32) theta[t] = __expf(theta_log[t]);
  if (t == 0) {
    float Aval = -__expf(A_log[0]);
    float run_dt = 0.f, run_la = 0.f;
    for (int l = 0; l < LSEQ; ++l) {
      float dtr = proj[(size_t)l * DPROJ + OFF_DT];
      float lmr = proj[(size_t)l * DPROJ + OFF_LAM];
      float dt  = (dtr > 20.f) ? dtr : log1pf(__expf(dtr));
      float lam = 1.f / (1.f + __expf(-lmr));
      run_dt += dt;
      csdt[l] = run_dt;
      if ((l & (CHUNK - 1)) == 0) run_la = 0.f;
      run_la += dt * Aval;
      lacs[l]   = run_la;
      uscale[l] = dt * lam;
      if ((l & (CHUNK - 1)) == (CHUNK - 1)) decayc[l >> 7] = __expf(run_la);
    }
  }
}

// =====================================================================
// Per-l: RMSNorm(B,C)+bias+RoPE  and  x = x' @ W_xup^T
// grid = L blocks of 128 threads
// =====================================================================
__global__ void prep_bcx(const float* __restrict__ proj,
                         const float* __restrict__ csdt, const float* __restrict__ theta,
                         const float* __restrict__ wnB, const float* __restrict__ wnC,
                         const float* __restrict__ biasB, const float* __restrict__ biasC,
                         const float* __restrict__ W_xup,
                         float* __restrict__ Brot, float* __restrict__ Crot,
                         float* __restrict__ xw) {
  const int l = blockIdx.x;
  const int tid = threadIdx.x;           // 128
  __shared__ float braw[128], craw[128], bp[128], cp[128], red[128];
  __shared__ float xp[1024];
  __shared__ float wx[128 * 64];

  const float* row = proj + (size_t)l * DPROJ;
  braw[tid] = row[OFF_B + tid];
  craw[tid] = row[OFF_C + tid];
  for (int i = tid; i < 1024; i += 128) xp[i] = row[OFF_X + i];
  for (int i = tid; i < 128 * 64; i += 128) wx[i] = W_xup[i];
  __syncthreads();

  red[tid] = braw[tid] * braw[tid]; __syncthreads();
  for (int s = 64; s > 0; s >>= 1) { if (tid < s) red[tid] += red[tid + s]; __syncthreads(); }
  float rb = rsqrtf(red[0] / 128.f + 1e-5f); __syncthreads();
  red[tid] = craw[tid] * craw[tid]; __syncthreads();
  for (int s = 64; s > 0; s >>= 1) { if (tid < s) red[tid] += red[tid + s]; __syncthreads(); }
  float rc = rsqrtf(red[0] / 128.f + 1e-5f); __syncthreads();

  bp[tid] = braw[tid] * rb * wnB[tid] + biasB[tid];
  cp[tid] = craw[tid] * rc * wnC[tid] + biasC[tid];
  __syncthreads();

  { // RoPE: flat index tid = n*2 + r ; pair n = n2*2 + cpos
    int n = tid >> 1, r = tid & 1;
    int n2 = n >> 1, cpos = n & 1;
    float ang = csdt[l] * theta[n2];
    float ca = __cosf(ang), sa = __sinf(ang);
    float re_b = bp[(n2 * 2 + 0) * 2 + r], im_b = bp[(n2 * 2 + 1) * 2 + r];
    float re_c = cp[(n2 * 2 + 0) * 2 + r], im_c = cp[(n2 * 2 + 1) * 2 + r];
    Brot[(size_t)l * 128 + tid] = cpos == 0 ? (re_b * ca - im_b * sa) : (re_b * sa + im_b * ca);
    Crot[(size_t)l * 128 + tid] = cpos == 0 ? (re_c * ca - im_c * sa) : (re_c * sa + im_c * ca);
  }

  // x up-projection: out[h][q] = sum_p x'[h][p] * W_xup[q][p]
  for (int idx = tid; idx < 2048; idx += 128) {
    int hh = idx >> 7, q = idx & 127;
    float s = 0.f;
    for (int p = 0; p < 64; ++p) s += xp[hh * 64 + p] * wx[q * 64 + p];
    xw[((size_t)l * HHEADS + hh) * 128 + q] = s;
  }
}

// =====================================================================
// Core: per (chunk c, head h) block.  TDM stages C/B/x tiles into LDS,
// h_intra = Lm @ u_c via WMMA, fused y_diag contraction, h_last extract.
// grid = 128 blocks (c*16+h), 256 threads, ~312KB dynamic LDS.
// =====================================================================
__global__ void chunk_core(const float* __restrict__ Brot, const float* __restrict__ Crot,
                           const float* __restrict__ xw,
                           const float* __restrict__ lacs, const float* __restrict__ uscale,
                           float* __restrict__ ydiag, float* __restrict__ hlast) {
  extern __shared__ char smem[];
  // No static LDS in this kernel -> dynamic region starts at byte offset 0,
  // so these offsets are also the D#.lds_addr values for the TDM.
  constexpr unsigned OFF_CSC = 0;               // f32[128*128] C_rot tile (TDM)
  constexpr unsigned OFF_BST = 65536;           // f32[128*128] B_rot tile (TDM)
  constexpr unsigned OFF_XST = 131072;          // f32[128*128] x tile     (TDM)
  constexpr unsigned OFF_YAC = 196608;          // f32[128*128] y_diag accumulator
  constexpr unsigned OFF_LM  = 262144;          // bf16[128*128] decay matrix
  constexpr unsigned OFF_UCS = 294912;          // bf16[128*64]  u_c K x N slab
  constexpr unsigned OFF_LAS = 311296;          // f32[128]
  constexpr unsigned OFF_USS = 311808;          // f32[128]
  float*  csC  = (float*)(smem + OFF_CSC);
  float*  bst  = (float*)(smem + OFF_BST);
  float*  xst  = (float*)(smem + OFF_XST);
  float*  yac  = (float*)(smem + OFF_YAC);
  __bf16* Lm   = (__bf16*)(smem + OFF_LM);
  __bf16* ucs  = (__bf16*)(smem + OFF_UCS);
  float*  la_s = (float*)(smem + OFF_LAS);
  float*  us_s = (float*)(smem + OFF_USS);

  const int c = blockIdx.x >> 4;
  const int h = blockIdx.x & 15;
  const int tid = threadIdx.x;             // 256
  const int l0 = c * CHUNK;

  // Wave 0 kicks off the three tensor DMAs; other waves do scalar staging
  // and accumulator init concurrently.
  if (tid < 32) {
    // C_rot rows l0..l0+127: contiguous 64KB (1 row x 16384 f32)
    tdm_load_2d_f32(OFF_CSC, Crot + (size_t)l0 * 128, 16384u, 1u, 16384ull, 16384u, 1u);
    // B_rot rows: same contiguous shape
    tdm_load_2d_f32(OFF_BST, Brot + (size_t)l0 * 128, 16384u, 1u, 16384ull, 16384u, 1u);
    // x tile: 128 rows of 128 f32, row stride H*128 = 2048 elements
    tdm_load_2d_f32(OFF_XST, xw + ((size_t)l0 * HHEADS + h) * 128, 128u, 128u,
                    2048ull, 128u, 1u << 20);
    __builtin_amdgcn_s_wait_tensorcnt(0);
  }
  for (int i = tid; i < 128; i += 256) { la_s[i] = lacs[l0 + i]; us_s[i] = uscale[l0 + i]; }
  for (int idx = tid; idx < 128 * 128; idx += 256) yac[idx] = 0.f;
  __syncthreads();

  for (int idx = tid; idx < 128 * 128; idx += 256) {
    int i = idx >> 7, j = idx & 127;
    Lm[idx] = (i >= j) ? f2bf(__expf(la_s[i] - la_s[j])) : f2bf(0.f);
  }
  __syncthreads();

  const int lane = tid & 31;
  const int wv   = tid >> 5;   // wave owns output rows [16*wv, 16*wv+16)
  const int mrow = lane & 15;
  const int kh   = lane >> 4;

  // Hoist the 4 A-fragments of Lm for this wave's M strip (reused over n loop)
  v16bf afrag[4];
#pragma unroll
  for (int kt = 0; kt < 4; ++kt)
#pragma unroll
    for (int s = 0; s < 16; ++s) {
      int kk = kt * 32 + ((s < 8) ? (kh * 8 + s) : (16 + kh * 8 + (s - 8)));
      afrag[kt][s] = Lm[(wv * 16 + mrow) * 128 + kk];
    }

  for (int n = 0; n < 64; ++n) {
    // u_c[j][p] = uscale[j] * sum_r B_rot[j,n,r] * x[j,h,p,r]
    for (int idx = tid; idx < 128 * 64; idx += 256) {
      int j = idx >> 6, p = idx & 63;
      float b0 = bst[j * 128 + n * 2 + 0];
      float b1 = bst[j * 128 + n * 2 + 1];
      float x0 = xst[j * 128 + p * 2 + 0];
      float x1 = xst[j * 128 + p * 2 + 1];
      ucs[idx] = f2bf(us_s[j] * (b0 * x0 + b1 * x1));
    }
    __syncthreads();

    v8f acc[4] = {};
#pragma unroll
    for (int pt = 0; pt < 4; ++pt) {
#pragma unroll
      for (int kt = 0; kt < 4; ++kt) {
        v16bf b;
#pragma unroll
        for (int s = 0; s < 16; ++s)
          b[s] = ucs[(kt * 32 + ((s < 8) ? (kh * 8 + s) : (16 + kh * 8 + (s - 8)))) * 64 + pt * 16 + (lane & 15)];
        acc[pt] = __builtin_amdgcn_wmma_f32_16x16x32_bf16(
            false, afrag[kt], false, b, (short)0, acc[pt], false, false);
      }
    }
    // y_diag[i,p,r] += h_intra[i,n,p] * C[i,n,r] ; h_last[n,p] = h_intra[127,n,p]
#pragma unroll
    for (int pt = 0; pt < 4; ++pt)
#pragma unroll
      for (int v = 0; v < 8; ++v) {
        int i = wv * 16 + v + 8 * kh;
        int p = pt * 16 + (lane & 15);
        float ht = acc[pt][v];
        float c0 = csC[i * 128 + n * 2 + 0];
        float c1 = csC[i * 128 + n * 2 + 1];
        yac[i * 128 + p * 2 + 0] += ht * c0;
        yac[i * 128 + p * 2 + 1] += ht * c1;
        if (i == 127) hlast[(((size_t)c * HHEADS + h) * 64 + n) * 64 + p] = ht;
      }
    __syncthreads();
  }

  for (int idx = tid; idx < 128 * 128; idx += 256) {
    int i = idx >> 7, q = idx & 127;
    ydiag[((size_t)(l0 + i) * HHEADS + h) * 128 + q] = yac[idx];
  }
}

// =====================================================================
// Inter-chunk sequential state scan (8 chunks).  grid = 16 (h), 256 thr
// =====================================================================
__global__ void chunk_scan(const float* __restrict__ hlast,
                           const float* __restrict__ decayc,
                           float* __restrict__ hprev) {
  int h = blockIdx.x;
  for (int idx = threadIdx.x; idx < 4096; idx += 256) {
    float state = 0.f;
    for (int c = 0; c < NCHUNK; ++c) {
      size_t base = ((size_t)c * HHEADS + h) * 4096 + idx;
      hprev[base] = state;
      state = state * decayc[c] + hlast[base];
    }
  }
}

// =====================================================================
// Per-l: y_off + y_diag, down-proj W_ydown, +D*x', SiLU(z) gate
// grid = L blocks of 128 threads
// =====================================================================
__global__ void combine_gate(const float* __restrict__ ydiag,
                             const float* __restrict__ hprev,
                             const float* __restrict__ Crot,
                             const float* __restrict__ lacs,
                             const float* __restrict__ proj,
                             const float* __restrict__ W_ydown,
                             const float* __restrict__ Dv,
                             float* __restrict__ ycore) {
  const int l = blockIdx.x;
  const int tid = threadIdx.x;           // 128
  const int c = l >> 7;
  __shared__ float Cl[128];
  __shared__ float yf[2048];
  Cl[tid] = Crot[(size_t)l * 128 + tid];
  __syncthreads();
  const float dec = __expf(lacs[l]);

  for (int idx = tid; idx < 1024; idx += 128) {
    int hh = idx >> 6, p = idx & 63;
    const float* hp = hprev + ((size_t)c * HHEADS + hh) * 4096 + p;
    float a0 = 0.f, a1 = 0.f;
    for (int n = 0; n < 64; ++n) {
      float hv = hp[n * 64];
      a0 += hv * Cl[n * 2 + 0];
      a1 += hv * Cl[n * 2 + 1];
    }
    size_t base = ((size_t)l * HHEADS + hh) * 128;
    yf[hh * 128 + p * 2 + 0] = ydiag[base + p * 2 + 0] + dec * a0;
    yf[hh * 128 + p * 2 + 1] = ydiag[base + p * 2 + 1] + dec * a1;
  }
  __syncthreads();

  const float* row = proj + (size_t)l * DPROJ;
  for (int idx = tid; idx < 1024; idx += 128) {
    int hh = idx >> 6, p = idx & 63;
    float s = 0.f;
    for (int q = 0; q < 128; ++q) s += yf[hh * 128 + q] * W_ydown[p * 128 + q];
    s += Dv[hh] * row[OFF_X + idx];
    float zv = row[OFF_Z + idx];
    float sig = 1.f / (1.f + __expf(-zv));
    ycore[(size_t)l * 1024 + idx] = s * (zv * sig);
  }
}

// =====================================================================
extern "C" void kernel_launch(void* const* d_in, const int* in_sizes, int n_in,
                              void* d_out, int out_size, void* d_ws, size_t ws_size,
                              hipStream_t stream) {
  const float* u         = (const float*)d_in[0];
  const float* W_in      = (const float*)d_in[1];
  const float* b_in      = (const float*)d_in[2];
  const float* W_xup     = (const float*)d_in[3];
  const float* W_ydown   = (const float*)d_in[4];
  const float* A_log     = (const float*)d_in[5];
  const float* theta_log = (const float*)d_in[6];
  const float* Dv        = (const float*)d_in[7];
  const float* w_normB   = (const float*)d_in[8];
  const float* w_normC   = (const float*)d_in[9];
  const float* bias_B    = (const float*)d_in[10];
  const float* bias_C    = (const float*)d_in[11];
  const float* W_out     = (const float*)d_in[12];
  float* out = (float*)d_out;

  char* ws = (char*)d_ws;
  float* proj   = (float*)(ws + WS_PROJ);
  float* csdt   = (float*)(ws + WS_CSDT);
  float* lacs   = (float*)(ws + WS_LACS);
  float* usc    = (float*)(ws + WS_USC);
  float* theta  = (float*)(ws + WS_THETA);
  float* decayc = (float*)(ws + WS_DECC);
  float* Brot   = (float*)(ws + WS_BROT);
  float* Crot   = (float*)(ws + WS_CROT);
  float* xwv    = (float*)(ws + WS_XW);
  float* ydiag  = (float*)(ws + WS_YDIAG);
  float* hlast  = (float*)(ws + WS_HLAST);
  float* hprev  = (float*)(ws + WS_HPREV);
  float* ycore  = (float*)(ws + WS_YCORE);

  // 1) proj = u @ W_in^T + b_in   (M=1024, N=2306, K=512)
  gemm_bf16_wmma<<<dim3((DPROJ + 127) / 128, LSEQ / 128), 256, 0, stream>>>(
      u, W_in, b_in, proj, LSEQ, DPROJ, DMODEL);

  // 2) scalar scans
  prep_scalars<<<1, 32, 0, stream>>>(proj, A_log, theta_log, csdt, lacs, usc, theta, decayc);

  // 3) RMSNorm+RoPE for B/C, x up-projection
  prep_bcx<<<LSEQ, 128, 0, stream>>>(proj, csdt, theta, w_normB, w_normC,
                                     bias_B, bias_C, W_xup, Brot, Crot, xwv);

  // 4) intra-chunk WMMA core with TDM-staged tiles (~312KB dynamic LDS)
  size_t smem_core = 312320;
  chunk_core<<<NCHUNK * HHEADS, 256, smem_core, stream>>>(
      Brot, Crot, xwv, lacs, usc, ydiag, hlast);

  // 5) inter-chunk scan
  chunk_scan<<<HHEADS, 256, 0, stream>>>(hlast, decayc, hprev);

  // 6) combine + down-proj + gate
  combine_gate<<<LSEQ, 128, 0, stream>>>(ydiag, hprev, Crot, lacs, proj,
                                         W_ydown, Dv, ycore);

  // 7) out = ycore @ W_out^T   (M=1024, N=512, K=1024)
  gemm_bf16_wmma<<<dim3(DMODEL / 128, LSEQ / 128), 256, 0, stream>>>(
      ycore, W_out, nullptr, out, LSEQ, DMODEL, 2 * DMODEL);
}
